// LSTM_52286931861890
// MI455X (gfx1250) — matrix operand order repaired and checked by reading, not compiled, not run on hardware
//
#include <hip/hip_runtime.h>

// ---------------------------------------------------------------------------
// LSTM scan, T=512, B=128, H=512.  Persistent-kernel design:
//   prep kernel : Wi/Wh f32 -> bf16 transposed WT[n][k]; zero h0; zero barrier.
//   scan kernel : 32 blocks x 256 threads (8 waves). Wave owns one 16x16
//     (batch x hidden) tile; c-tile lives in VGPRs across all 512 steps.
//     Per step: stage x_t (f32->bf16) and reset-masked h into LDS,
//     128x v_wmma_f32_16x16x32_bf16 (4 gates x 2 GEMMs x K=512/32),
//     branch-free gate nonlinearities (v_exp/v_rcp), write ys + h(bf16),
//     device-wide barrier (monotonic atomic counter, reset each launch).
// ---------------------------------------------------------------------------

typedef __attribute__((ext_vector_type(16))) __bf16 v16bf;
typedef __attribute__((ext_vector_type(8)))  float  v8f;

union BF16x16 {
  v16bf v;
  uint4 q[2];
};

__device__ __forceinline__ unsigned short f2bf(float f) {
  unsigned int u = __float_as_uint(f);
  u += 0x7FFFu + ((u >> 16) & 1u);        // round-to-nearest-even
  return (unsigned short)(u >> 16);
}

__device__ __forceinline__ unsigned int pack2bf(float lo, float hi) {
  return (unsigned int)f2bf(lo) | ((unsigned int)f2bf(hi) << 16);
}

// Branch-free sigmoid: v_exp_f32 + v_rcp_f32.
__device__ __forceinline__ float sigmoidf_fast(float x) {
  return __builtin_amdgcn_rcpf(1.0f + __expf(-x));
}

// Branch-free tanh: e = exp(-2|x|) in (0,1]; tanh = sign(x)*(1-e)/(1+e).
__device__ __forceinline__ float tanhf_fast(float x) {
  const float e = __expf(-2.0f * __builtin_fabsf(x));
  const float r = (1.0f - e) * __builtin_amdgcn_rcpf(1.0f + e);
  return __builtin_copysignf(r, x);
}

#define WMMA_BF16(A, Bm, C) \
  __builtin_amdgcn_wmma_f32_16x16x32_bf16(false, (A), false, (Bm), (short)0, \
                                          (C), false, false)

// --------------------------- prep kernel -----------------------------------
__global__ void __launch_bounds__(256)
lstm_prep_kernel(const float* __restrict__ Wi, const float* __restrict__ Wh,
                 unsigned short* __restrict__ wtI,
                 unsigned short* __restrict__ wtH,
                 unsigned short* __restrict__ hglob,
                 unsigned int* __restrict__ ctr) {
  const unsigned int gid = blockIdx.x * 256u + threadIdx.x;
  if (gid < (1u << 20)) {                 // 2048*512 elements
    const unsigned int n = gid >> 9;      // output column 0..2047
    const unsigned int k = gid & 511u;    // K index       0..511
    wtI[gid] = f2bf(Wi[(size_t)k * 2048 + n]);   // WT[n][k] = W[k][n]
    wtH[gid] = f2bf(Wh[(size_t)k * 2048 + n]);
  }
  if (gid < 65536u) hglob[gid] = 0;       // h0 = 0 (bf16)
  if (gid == 0u) *ctr = 0u;               // barrier counter reset each launch
}

// --------------------------- scan kernel -----------------------------------
__global__ void __launch_bounds__(256, 1)
lstm_scan_kernel(const float* __restrict__ ins,
                 const unsigned char* __restrict__ resets,
                 const float* __restrict__ bvec,
                 const unsigned short* __restrict__ wtI,
                 const unsigned short* __restrict__ wtH,
                 unsigned short* __restrict__ hglob,
                 unsigned int* __restrict__ barrier_ctr,
                 float* __restrict__ out) {
  __shared__ unsigned short x_lds[16 * 512];   // 16 rows x K=512 bf16 (16 KB)
  __shared__ unsigned short h_lds[16 * 512];

  const int tid  = threadIdx.x;
  const int wave = tid >> 5;
  const int lane = tid & 31;
  const int rb   = blockIdx.x >> 2;                    // batch row block 0..7
  const int cb   = ((blockIdx.x & 3) << 3) + wave;     // hidden col block 0..31
  const int m    = lane & 15;
  const int sel  = lane >> 4;
  const int ncol = cb * 16 + m;                        // this lane's N column

  const float bias0 = bvec[ncol];
  const float bias1 = bvec[512 + ncol];
  const float bias2 = bvec[1024 + ncol];
  const float bias3 = bvec[1536 + ncol];

  // One base pointer per weight matrix; gate offset (g*512*512 elements)
  // folds into the 24-bit instruction immediate of global_load_b128.
  const unsigned short* wbI = wtI + (size_t)ncol * 512 + (size_t)(sel * 16);
  const unsigned short* wbH = wtH + (size_t)ncol * 512 + (size_t)(sel * 16);

  const int a_off = m * 512 + sel * 16;  // A-fragment LDS base for this lane

  float cst[8];                          // persistent c tile (acc layout)
#pragma unroll
  for (int r = 0; r < 8; ++r) cst[r] = 0.0f;

#pragma unroll 1
  for (int t = 0; t < 512; ++t) {
    const size_t rbase = (size_t)t * 128 + rb * 16;

    // ---- stage x_t (f32 -> bf16, float4 granularity) ---------------------
    const float* xsrc = ins + ((size_t)t * 128 + rb * 16) * 512;
#pragma unroll 1
    for (int i = tid * 4; i < 8192; i += 1024) {
      const float4 xv = *(const float4*)(xsrc + i);
      *(uint2*)&x_lds[i] =
          make_uint2(pack2bf(xv.x, xv.y), pack2bf(xv.z, xv.w));
    }
    // ---- stage h (bf16 copy, reset rows zeroed) --------------------------
    const unsigned short* hsrc = hglob + (size_t)(rb * 16) * 512;
#pragma unroll 1
    for (int i = tid * 8; i < 8192; i += 2048) {
      uint4 hv = *(const uint4*)(hsrc + i);
      if (resets[rbase + (i >> 9)]) hv = make_uint4(0u, 0u, 0u, 0u);
      *(uint4*)&h_lds[i] = hv;
    }
    if (t + 1 < 512)                     // pull next step's x toward L2/L0
      __builtin_prefetch(xsrc + 65536 + tid * 32, 0, 0);
    __syncthreads();

    // c-reset mask bytes for this lane's 8 rows (aligned u64 read)
    const unsigned long long rbits =
        *(const unsigned long long*)(resets + rbase + sel * 8);

    // ---- z = x@Wi + h@Wh (bias added in epilogue) ------------------------
    v8f acc0 = (v8f)0.0f;
    v8f acc1 = (v8f)0.0f;
    v8f acc2 = (v8f)0.0f;
    v8f acc3 = (v8f)0.0f;

#pragma unroll 1
    for (int k0 = 0; k0 < 512; k0 += 32) {
      BF16x16 ax, ah;
      ax.q[0] = *(const uint4*)&x_lds[a_off + k0];
      ax.q[1] = *(const uint4*)&x_lds[a_off + k0 + 8];
      ah.q[0] = *(const uint4*)&h_lds[a_off + k0];
      ah.q[1] = *(const uint4*)&h_lds[a_off + k0 + 8];

      BF16x16 bw;
      // gate 0 (i)
      bw.q[0] = *(const uint4*)(wbI + k0);
      bw.q[1] = *(const uint4*)(wbI + k0 + 8);
      acc0 = WMMA_BF16(ax.v, bw.v, acc0);
      bw.q[0] = *(const uint4*)(wbH + k0);
      bw.q[1] = *(const uint4*)(wbH + k0 + 8);
      acc0 = WMMA_BF16(ah.v, bw.v, acc0);
      // gate 1 (f)
      bw.q[0] = *(const uint4*)(wbI + 262144 + k0);
      bw.q[1] = *(const uint4*)(wbI + 262144 + k0 + 8);
      acc1 = WMMA_BF16(ax.v, bw.v, acc1);
      bw.q[0] = *(const uint4*)(wbH + 262144 + k0);
      bw.q[1] = *(const uint4*)(wbH + 262144 + k0 + 8);
      acc1 = WMMA_BF16(ah.v, bw.v, acc1);
      // gate 2 (g)
      bw.q[0] = *(const uint4*)(wbI + 524288 + k0);
      bw.q[1] = *(const uint4*)(wbI + 524288 + k0 + 8);
      acc2 = WMMA_BF16(ax.v, bw.v, acc2);
      bw.q[0] = *(const uint4*)(wbH + 524288 + k0);
      bw.q[1] = *(const uint4*)(wbH + 524288 + k0 + 8);
      acc2 = WMMA_BF16(ah.v, bw.v, acc2);
      // gate 3 (o)
      bw.q[0] = *(const uint4*)(wbI + 786432 + k0);
      bw.q[1] = *(const uint4*)(wbI + 786432 + k0 + 8);
      acc3 = WMMA_BF16(ax.v, bw.v, acc3);
      bw.q[0] = *(const uint4*)(wbH + 786432 + k0);
      bw.q[1] = *(const uint4*)(wbH + 786432 + k0 + 8);
      acc3 = WMMA_BF16(ah.v, bw.v, acc3);
    }

    // ---- gates, state update, outputs (branch-free nonlinearities) -------
    float* ys = out + 131072 + (size_t)t * 65536;
#pragma unroll
    for (int r = 0; r < 8; ++r) {
      const float keep = ((rbits >> (8 * r)) & 0xFFull) ? 0.0f : 1.0f;
      const float cm = cst[r] * keep;
      const float iv = sigmoidf_fast(acc0[r] + bias0);
      const float fv = sigmoidf_fast(acc1[r] + bias1);
      const float gv = tanhf_fast(acc2[r] + bias2);
      const float ov = sigmoidf_fast(acc3[r] + bias3);
      const float nc = fv * cm + iv * gv;
      const float nh = ov * tanhf_fast(nc);
      cst[r] = nc;
      const int    mrow = rb * 16 + r + sel * 8;   // C/D layout row
      const size_t idx  = (size_t)mrow * 512 + ncol;
      ys[idx]    = nh;
      hglob[idx] = f2bf(nh);
      if (t == 511) {
        out[idx]         = nc;   // cT
        out[65536 + idx] = nh;   // hT
      }
    }

    // ---- device-wide barrier (monotonic counter, zeroed by prep) ---------
    __threadfence();
    __syncthreads();
    if (tid == 0) {
      atomicAdd(barrier_ctr, 1u);
      const unsigned int target = 32u * (unsigned int)(t + 1);
      while (__hip_atomic_load(barrier_ctr, __ATOMIC_RELAXED,
                               __HIP_MEMORY_SCOPE_AGENT) < target) {
        __builtin_amdgcn_s_sleep(2);
      }
    }
    __syncthreads();
    __threadfence();
  }
}

// --------------------------- launcher --------------------------------------
extern "C" void kernel_launch(void* const* d_in, const int* in_sizes, int n_in,
                              void* d_out, int out_size, void* d_ws,
                              size_t ws_size, hipStream_t stream) {
  (void)in_sizes; (void)n_in; (void)out_size; (void)ws_size;
  const float*         ins    = (const float*)d_in[0];
  const unsigned char* resets = (const unsigned char*)d_in[1];  // jax bool, 1B
  const float*         Wi     = (const float*)d_in[2];
  const float*         Wh     = (const float*)d_in[3];
  const float*         b      = (const float*)d_in[4];
  float* out = (float*)d_out;

  char* ws = (char*)d_ws;
  unsigned short* wtI = (unsigned short*)(ws);                     // 2 MB
  unsigned short* wtH = (unsigned short*)(ws + (1u << 21));        // 2 MB
  unsigned short* hg  = (unsigned short*)(ws + (1u << 22));        // 128 KB
  unsigned int*   ctr = (unsigned int*)(ws + (1u << 22) + (1u << 17));

  lstm_prep_kernel<<<4096, 256, 0, stream>>>(Wi, Wh, wtI, wtH, hg, ctr);
  lstm_scan_kernel<<<32, 256, 0, stream>>>(ins, resets, b, wtI, wtH, hg, ctr,
                                           out);
}
